// TemporalML_74698071212200
// MI455X (gfx1250) — matrix operand order repaired and checked by reading, not compiled, not run on hardware
//
#include <hip/hip_runtime.h>

#define N_NODES 2000
#define NPAD    2016
#define EMB     64
#define TD      384
#define T_LOOK  24
#define IN_DIM  16
#define BATCH   32

typedef __attribute__((ext_vector_type(8)))  float        v8f;
typedef __attribute__((ext_vector_type(8)))  unsigned int v8u;
typedef __attribute__((ext_vector_type(16))) __bf16       v16bf;

static __device__ __forceinline__ unsigned short f2bf(float f) {
  unsigned int u = __float_as_uint(f);
  u += 0x7FFFu + ((u >> 16) & 1u);   // round-to-nearest-even
  return (unsigned short)(u >> 16);
}

// pack two floats to packed bf16x2 (RNE)
static __device__ __forceinline__ unsigned int pack_bf16(float a, float b) {
  unsigned int ua = __float_as_uint(a);
  unsigned int ub = __float_as_uint(b);
  ua += 0x7FFFu + ((ua >> 16) & 1u);
  ub += 0x7FFFu + ((ub >> 16) & 1u);
  return (ua >> 16) | (ub & 0xFFFF0000u);
}

static __device__ __forceinline__ v8u pack2(uint4 a, uint4 b) {
  v8u r;
  r[0] = a.x; r[1] = a.y; r[2] = a.z; r[3] = a.w;
  r[4] = b.x; r[5] = b.y; r[6] = b.z; r[7] = b.w;
  return r;
}

static __device__ __forceinline__ v8f wmma_bf16(v8u a, v8u b, v8f c) {
  return __builtin_amdgcn_wmma_f32_16x16x32_bf16(
      false, __builtin_bit_cast(v16bf, a),
      false, __builtin_bit_cast(v16bf, b),
      (short)0, c, false, false);
}

// ---- VALU cross-lane exchanges (v_permlane16_b32 family; no LDS, no waits) ----
static __device__ __forceinline__ float plx(float v, unsigned s1, unsigned s2) {
  int s = __float_as_int(v);
  return __int_as_float(__builtin_amdgcn_permlane16(s, s, s1, s2, false, false));
}
static __device__ __forceinline__ float xor1_f (float v) { return plx(v, 0x67452301u, 0xEFCDAB89u); }
static __device__ __forceinline__ float xor2_f (float v) { return plx(v, 0x54761032u, 0xDCFE98BAu); }
static __device__ __forceinline__ float xor4_f (float v) { return plx(v, 0x32107654u, 0xBA98FEDCu); }
static __device__ __forceinline__ float xor8_f (float v) { return plx(v, 0xFEDCBA98u, 0x76543210u); }
static __device__ __forceinline__ float xor16_f(float v) {
  int s = __float_as_int(v);
  return __int_as_float(
      __builtin_amdgcn_permlanex16(s, s, 0x76543210u, 0xFEDCBA98u, false, false));
}
// full-row (16-lane) sum; every lane in the row ends with the total
static __device__ __forceinline__ float rowsum16(float v) {
  v += xor1_f(v); v += xor2_f(v); v += xor4_f(v); v += xor8_f(v);
  return v;
}
// full-wave (32-lane) sum
static __device__ __forceinline__ float wavesum32(float v) {
  v = rowsum16(v); v += xor16_f(v);
  return v;
}

// ---------------------------------------------------------------------------
// Phase 1: q1/k1 = layernorm(x @ W + b) in bf16.  One wave32 per (b,n) row.
// ---------------------------------------------------------------------------
static __device__ __forceinline__ void ln_store(
    float a0, float a1, const float* g, const float* be,
    unsigned short* dst, int e0, int e1)
{
  float s = wavesum32(a0 + a1);
  float mu = s * (1.0f / 64.0f);
  float d0 = a0 - mu, d1 = a1 - mu;
  float v = wavesum32(d0 * d0 + d1 * d1);
  float rstd = rsqrtf(v * (1.0f / 64.0f) + 1e-5f);
  dst[e0] = f2bf(d0 * rstd * g[e0] + be[e0]);
  dst[e1] = f2bf(d1 * rstd * g[e1] + be[e1]);
}

__global__ __launch_bounds__(256) void qk_kernel(
    const float* __restrict__ x,
    const float* __restrict__ Wq, const float* __restrict__ bq,
    const float* __restrict__ Wk, const float* __restrict__ bk,
    const float* __restrict__ g0, const float* __restrict__ beta0,
    const float* __restrict__ g1, const float* __restrict__ beta1,
    unsigned short* __restrict__ q1, unsigned short* __restrict__ k1)
{
  const int wave = threadIdx.x >> 5;
  const int lane = threadIdx.x & 31;
  const size_t row = (size_t)blockIdx.x * 8 + wave;       // < 64000
  const int b = (int)(row / N_NODES);
  const int n = (int)(row % N_NODES);
  const float* xr = x + row * TD;
  const int e0 = lane, e1 = lane + 32;

  float aq0 = bq[e0], aq1 = bq[e1], ak0 = bk[e0], ak1 = bk[e1];
  for (int k = 0; k < TD; ++k) {
    float xv = xr[k];                                     // uniform → broadcast
    aq0 = fmaf(xv, Wq[k * EMB + e0], aq0);
    aq1 = fmaf(xv, Wq[k * EMB + e1], aq1);
    ak0 = fmaf(xv, Wk[k * EMB + e0], ak0);
    ak1 = fmaf(xv, Wk[k * EMB + e1], ak1);
  }
  ln_store(aq0, aq1, g0, beta0, q1 + row * EMB, e0, e1);
  ln_store(ak0, ak1, g1, beta1,
           k1 + ((size_t)b * NPAD + n) * EMB, e0, e1);
}

// ---------------------------------------------------------------------------
// Phase 1b: V^T (feature-major bf16, m padded to 2016 with zeros) + k1 pad=0
// ---------------------------------------------------------------------------
__global__ __launch_bounds__(256) void prep_kernel(
    const float* __restrict__ normal,
    unsigned short* __restrict__ vt, unsigned short* __restrict__ k1)
{
  size_t i = (size_t)blockIdx.x * 256 + threadIdx.x;
  const size_t VTN = (size_t)TD * NPAD;
  if (i < VTN) {
    size_t f = i / NPAD, m = i % NPAD;
    vt[i] = (m < N_NODES) ? f2bf(normal[m * TD + f]) : (unsigned short)0;
  } else if (i < VTN + (size_t)BATCH * 16 * EMB) {
    size_t j = i - VTN;
    size_t bb = j / (16 * EMB), r = j % (16 * EMB);
    k1[(bb * NPAD + N_NODES) * EMB + r] = 0;
  }
}

// ---------------------------------------------------------------------------
// Phase 2: fused attention.  S^T = K_tile · Q1^T so the D-tile layout
// (lane = q-row, slots = m) is exactly the A-operand layout for the following
// P·V WMMA -> probabilities repacked to bf16 entirely in registers (no LDS,
// no shuffles in the hot loop).  Softmax denominator accumulates as a 13th
// tile O_sum = P · Ones (constant B operand).  Scores are bounded
// (|q·k|/sqrt(2000) <= ~1.5 for layernormed rows) so plain exp is safe.
// ---------------------------------------------------------------------------
__global__ __launch_bounds__(64) void attn_kernel(
    const unsigned short* __restrict__ q1,   // [B][2000][64] bf16
    const unsigned short* __restrict__ k1,   // [B][2016][64] bf16
    const unsigned short* __restrict__ vt,   // [384][2016]   bf16
    const float* __restrict__ x,             // [B][2000][24][16]
    const float* __restrict__ Win,           // [16][16]
    float* __restrict__ out)                 // [B][2000][24][16]
{
  const int lane = threadIdx.x & 31;
  const int wave = threadIdx.x >> 5;
  const int lh   = lane & 15;
  const int hi   = lane >> 4;
  const int hi8  = hi * 8;
  const int n0   = blockIdx.x * 16;
  const int b    = blockIdx.y;

  __shared__ float ldsAtt[16][T_LOOK];

  // Q1^T as resident B-operand (column = q-row lh, K = e in table order: hi8+{0..7}, 16+hi8+{0..7}).
  const unsigned short* qrow = q1 + ((size_t)b * N_NODES + n0 + lh) * EMB;
  v8u Qb[2];
#pragma unroll
  for (int kk = 0; kk < 2; ++kk) {
    const unsigned short* p = qrow + kk * 32 + hi8;
    Qb[kk] = pack2(*(const uint4*)p, *(const uint4*)(p + 16));
  }

  v8f O[13];                                  // 12 feature tiles + row-sum tile
#pragma unroll
  for (int j = 0; j < 13; ++j)
#pragma unroll
    for (int i = 0; i < 8; ++i) O[j][i] = 0.0f;

  v8u OnesB;                                  // bf16 1.0 everywhere
#pragma unroll
  for (int e = 0; e < 8; ++e) OnesB[e] = 0x3F803F80u;

  const unsigned short* kb = k1 + (size_t)b * NPAD * EMB;
  const float scale = 0.02236067977499790f;   // 1/sqrt(2000)

  for (int mc = 0; mc < NPAD; mc += 32) {
    // ---- S^T strip: two 16m x 16q tiles via 4 bf16 WMMAs ----
    v8f St0, St1;
#pragma unroll
    for (int i = 0; i < 8; ++i) { St0[i] = 0.0f; St1[i] = 0.0f; }
#pragma unroll
    for (int kk = 0; kk < 2; ++kk) {
      const unsigned short* a0 = kb + (size_t)(mc + lh) * EMB + kk * 32 + hi8;
      const unsigned short* a1 = kb + (size_t)(mc + 16 + lh) * EMB + kk * 32 + hi8;
      v8u Ka0 = pack2(*(const uint4*)a0, *(const uint4*)(a0 + 16));
      v8u Ka1 = pack2(*(const uint4*)a1, *(const uint4*)(a1 + 16));
      St0 = wmma_bf16(Ka0, Qb[kk], St0);
      St1 = wmma_bf16(Ka1, Qb[kk], St1);
    }

    // ---- p = exp(s/sqrt(N)); mask padded m rows (only last chunk, tile 1) ----
    float p0[8], p1[8];
#pragma unroll
    for (int i = 0; i < 8; ++i) {
      p0[i] = __expf(St0[i] * scale);                       // m = mc+i+hi8 < 2000 always
      p1[i] = (mc + 16 + i + hi8 >= N_NODES) ? 0.0f
                                             : __expf(St1[i] * scale);
    }

    // ---- in-register repack: S^T D-layout == P A-operand layout ----
    v8u Pa;
#pragma unroll
    for (int j = 0; j < 4; ++j) {
      Pa[j]     = pack_bf16(p0[2 * j], p0[2 * j + 1]);      // K = hi8 + {0..7}
      Pa[4 + j] = pack_bf16(p1[2 * j], p1[2 * j + 1]);      // K = 16 + hi8 + {0..7}
    }

    // ---- O += P * V (12 tiles) and O_sum += P * Ones ----
#pragma unroll
    for (int j = 0; j < 12; ++j) {
      const unsigned short* vr =
          vt + (size_t)(wave * 192 + j * 16 + lh) * NPAD + mc + hi8;
      v8u Vb = pack2(*(const uint4*)vr, *(const uint4*)(vr + 16));
      O[j] = wmma_bf16(Pa, Vb, O[j]);
    }
    O[12] = wmma_bf16(Pa, OnesB, O[12]);
  }

  // ---- normalize by softmax denominator (row sums live in O[12] slots) ----
  float invl[8];
#pragma unroll
  for (int i = 0; i < 8; ++i) invl[i] = 1.0f / O[12][i];
#pragma unroll
  for (int j = 0; j < 12; ++j)
#pragma unroll
    for (int i = 0; i < 8; ++i) O[j][i] *= invl[i];

  // ---- epilogue: query = x[:,:,-1,:] @ Win ; att over t ; out = data + w ----
  float qv[8];
#pragma unroll
  for (int i = 0; i < 8; ++i) {
    const float* xr =
        x + (((size_t)b * N_NODES + n0 + i + hi8) * T_LOOK + (T_LOOK - 1)) * IN_DIM;
    float a = 0.0f;
#pragma unroll
    for (int e = 0; e < IN_DIM; ++e) a = fmaf(xr[e], Win[e * IN_DIM + lh], a);
    qv[i] = a;
  }
#pragma unroll
  for (int j = 0; j < 12; ++j) {
    int t = wave * 12 + j;                 // column tile j == time step t
#pragma unroll
    for (int i = 0; i < 8; ++i) {
      float v = rowsum16(qv[i] * O[j][i]); // d = lane&15 within tile
      if (lh == 0) ldsAtt[i + hi8][t] = v;
    }
  }
  __syncthreads();

#pragma unroll
  for (int i = 0; i < 8; ++i) {
    int row = i + hi8;
    float wm = -3.0e38f;
#pragma unroll
    for (int t = 0; t < T_LOOK; ++t) wm = fmaxf(wm, ldsAtt[row][t]);
    float wsum = 0.0f;
    float we[T_LOOK];
#pragma unroll
    for (int t = 0; t < T_LOOK; ++t) { we[t] = __expf(ldsAtt[row][t] - wm); wsum += we[t]; }
    float inv = 1.0f / wsum;
    float* orow = out + (((size_t)b * N_NODES + n0 + row) * T_LOOK) * IN_DIM;
#pragma unroll
    for (int j = 0; j < 12; ++j) {
      int t = wave * 12 + j;
      orow[t * IN_DIM + lh] = O[j][i] + we[t] * inv;
    }
  }
}

// ---------------------------------------------------------------------------
extern "C" void kernel_launch(void* const* d_in, const int* in_sizes, int n_in,
                              void* d_out, int out_size, void* d_ws, size_t ws_size,
                              hipStream_t stream) {
  const float* x      = (const float*)d_in[0];
  const float* Wq     = (const float*)d_in[1];
  const float* bq     = (const float*)d_in[2];
  const float* Wk     = (const float*)d_in[3];
  const float* bk     = (const float*)d_in[4];
  const float* g0     = (const float*)d_in[5];
  const float* beta0  = (const float*)d_in[6];
  const float* g1     = (const float*)d_in[7];
  const float* beta1  = (const float*)d_in[8];
  const float* normal = (const float*)d_in[9];
  const float* Win    = (const float*)d_in[10];
  float* out = (float*)d_out;

  char* ws = (char*)d_ws;
  const size_t Q1B = (size_t)BATCH * N_NODES * EMB * 2;   // 8,192,000
  const size_t K1B = (size_t)BATCH * NPAD * EMB * 2;      // 8,257,536
  unsigned short* q1 = (unsigned short*)ws;
  unsigned short* k1 = (unsigned short*)(ws + Q1B);
  unsigned short* vt = (unsigned short*)(ws + Q1B + K1B);

  qk_kernel<<<(BATCH * N_NODES) / 8, 256, 0, stream>>>(
      x, Wq, bq, Wk, bk, g0, beta0, g1, beta1, q1, k1);

  size_t prep_n = (size_t)TD * NPAD + (size_t)BATCH * 16 * EMB;
  prep_kernel<<<(int)((prep_n + 255) / 256), 256, 0, stream>>>(normal, vt, k1);

  attn_kernel<<<dim3(N_NODES / 16, BATCH), 64, 0, stream>>>(
      q1, k1, vt, x, Win, out);
}
